// SelfAttention_75883482186327
// MI455X (gfx1250) — matrix operand order repaired and checked by reading, not compiled, not run on hardware
//
#include <hip/hip_runtime.h>
#include <hip/hip_bf16.h>

// Problem constants (match reference)
constexpr int Bn  = 4;
constexpr int Cc  = 256;
constexpr int QKd = 64;
constexpr int Nn  = 64 * 64;   // 4096
constexpr int KP  = 256;       // K-panel depth staged per TDM transfer

typedef __attribute__((ext_vector_type(16))) __bf16 v16bf;
typedef __attribute__((ext_vector_type(8)))  float  v8f;
typedef unsigned int u32x4 __attribute__((ext_vector_type(4)));
typedef int          i32x4 __attribute__((ext_vector_type(4)));
typedef int          i32x8 __attribute__((ext_vector_type(8)));

// ---------- helpers ----------

static __device__ inline __bf16 f2bf(float f) {
    unsigned u = __builtin_bit_cast(unsigned, f);
    unsigned r = (u + 0x7FFFu + ((u >> 16) & 1u)) >> 16;   // round-to-nearest-even
    unsigned short s = (unsigned short)r;
    return __builtin_bit_cast(__bf16, s);
}

// Load a 16x32 bf16 A-tile (rows m0..m0+15, cols k0..k0+31) from row-major A[lda].
// 16-bit A layout (contiguous K): lane l -> row (l&15); K pairs: VGPR v -> K=(hi*16)+2v{,+1},
// hi = l>>4. Pairs are contiguous -> dword loads.
static __device__ inline v16bf load_a_tile(const __bf16* __restrict__ A, int lda,
                                           int m0, int k0, int lane) {
    int row = m0 + (lane & 15);
    int hi8 = (lane >> 4) << 4;
    const __bf16* base = A + (long)row * lda + k0;
    union { v16bf v; unsigned u[8]; } t;
#pragma unroll
    for (int r = 0; r < 8; ++r) {
        int kv = hi8 + 2 * r;
        t.u[r] = *(const unsigned*)(base + kv);
    }
    return t.v;
}

// Load a 32x16 bf16 B-tile (rows k0..k0+31, cols n0..n0+15) from row-major B[ldb] (global).
// Contiguous-K B layout (matches load_a_tile): lane l -> col (l&15); lanes 0-15 hold K=0..15, lanes 16-31 K=16..31.
static __device__ inline v16bf load_b_tile(const __bf16* __restrict__ B, int ldb,
                                           int k0, int n0, int lane) {
    int col = n0 + (lane & 15);
    int kb  = k0 + ((lane >> 4) << 4);
    union { v16bf v; __bf16 e[16]; } t;
#pragma unroll
    for (int r = 0; r < 16; ++r) t.e[r] = B[(long)(kb + r) * ldb + col];
    return t.v;
}

// Same B-layout gather but from an LDS-resident [KP x 16] row-major panel.
static __device__ inline v16bf load_b_panel_lds(const __bf16* panel, int kk, int lane) {
    int col = lane & 15;
    int kb  = kk + ((lane >> 4) << 4);
    union { v16bf v; __bf16 e[16]; } t;
#pragma unroll
    for (int r = 0; r < 16; ++r) t.e[r] = panel[(kb + r) * 16 + col];
    return t.v;
}

// ---------- Tensor Data Mover: DMA a 2D bf16 tile (tile_d0 x tile_d1 rows) to LDS ----------
// D# per CDNA5 ISA ch.8: group0 = {count, lds_addr, global_addr, type=2},
// group1 = {data_size=2B, tensor dims, tile dims, dim0 stride}; groups 2/3 zero (2-D).
static __device__ inline void tdm_load_2d_to_lds(unsigned lds_addr, const void* gaddr,
                                                 unsigned tile_d0, unsigned tile_d1,
                                                 unsigned stride0_elems) {
    unsigned long long ga = (unsigned long long)(uintptr_t)gaddr;
    u32x4 g0;
    g0[0] = 1u;                                              // count=1 (valid user D#)
    g0[1] = lds_addr;                                        // LDS byte address
    g0[2] = (unsigned)(ga & 0xFFFFFFFFu);                    // global_addr[31:0]
    g0[3] = (unsigned)((ga >> 32) & 0x1FFFFFFu) | (2u << 30); // global_addr[56:32] | type=2
    i32x8 g1;
    g1[0] = (int)(1u << 16);                                 // data_size = 1 -> 2 bytes
    g1[1] = (int)((tile_d0 & 0xFFFFu) << 16);                // tensor_dim0[15:0] (== tile_d0)
    g1[2] = (int)((tile_d0 >> 16) | ((tile_d1 & 0xFFFFu) << 16)); // dim0 hi | tensor_dim1 lo
    g1[3] = (int)((tile_d1 >> 16) | ((tile_d0 & 0xFFFFu) << 16)); // dim1 hi | tile_dim0
    g1[4] = (int)(tile_d1 & 0xFFFFu);                        // tile_dim1 | tile_dim2=0
    g1[5] = (int)stride0_elems;                              // tensor_dim0_stride[31:0]
    g1[6] = 0;
    g1[7] = 0;
    i32x4 g2 = {0, 0, 0, 0};
    i32x4 g3 = {0, 0, 0, 0};
#if defined(__clang_major__) && (__clang_major__ >= 23)
    i32x8 g1b = {0, 0, 0, 0, 0, 0, 0, 0};
    __builtin_amdgcn_tensor_load_to_lds(g0, g1, g2, g3, g1b, 0);
#else
    __builtin_amdgcn_tensor_load_to_lds(g0, g1, g2, g3, 0);
#endif
}

// ---------- conversion kernels ----------

__global__ void convert_features_kernel(const float* __restrict__ f,
                                        __bf16* __restrict__ x_bf,
                                        __bf16* __restrict__ xT_bf) {
    long idx = (long)blockIdx.x * blockDim.x + threadIdx.x;
    long total = (long)Bn * Cc * Nn;
    if (idx >= total) return;
    long n = idx % Nn;
    long t = idx / Nn;
    long c = t % Cc;
    long b = t / Cc;
    __bf16 v = f2bf(f[idx]);
    x_bf[idx] = v;
    xT_bf[(b * Nn + n) * Cc + c] = v;
}

__global__ void convert_f32_bf16_kernel(const float* __restrict__ src,
                                        __bf16* __restrict__ dst, int n) {
    int i = blockIdx.x * blockDim.x + threadIdx.x;
    if (i < n) dst[i] = f2bf(src[i]);
}

// ---------- generic batched bf16 WMMA GEMM with TDM-staged B panels ----------
// D(b) = A(b)[M,K] * B(b)[K,N] (+ bias[row]); MODE 0: f32 row-major,
// MODE 1: bf16 row-major, MODE 2: bf16 transposed (stores D^T, ldd = row stride of D^T).
// All 4 waves of a block share n0, so the [KP x 16] B panel is staged once into LDS by
// the Tensor Data Mover (double-buffered), overlapping DMA with WMMA on the prior panel.
template <int MODE>
__global__ __launch_bounds__(128)
void gemm_bf16_wmma(const __bf16* __restrict__ A, long sA,
                    const __bf16* __restrict__ Bm, long sB,
                    const float* __restrict__ bias,
                    void* __restrict__ D, long sD,
                    int K, int lda, int ldb, int ldd) {
    __shared__ __bf16 Bpanel[2][KP * 16];   // 2 x 8 KB

    int lane = threadIdx.x & 31;
    int wv   = threadIdx.x >> 5;
    int b    = blockIdx.z;
    int n0   = blockIdx.x * 16;
    int m0   = (blockIdx.y * 4 + wv) * 16;
    const __bf16* Ab = A  + (long)b * sA;
    const __bf16* Bb = Bm + (long)b * sB + n0;   // fold column offset into the tile base

    int nP = K / KP;
    if (wv == 0)
        tdm_load_2d_to_lds((unsigned)(uintptr_t)&Bpanel[0][0], Bb, 16u, (unsigned)KP,
                           (unsigned)ldb);
    __builtin_amdgcn_s_wait_tensorcnt(0);
    __syncthreads();

    v8f acc = {0.f, 0.f, 0.f, 0.f, 0.f, 0.f, 0.f, 0.f};
    for (int p = 0; p < nP; ++p) {
        if (wv == 0 && p + 1 < nP)
            tdm_load_2d_to_lds((unsigned)(uintptr_t)&Bpanel[(p + 1) & 1][0],
                               Bb + (long)(p + 1) * KP * ldb, 16u, (unsigned)KP,
                               (unsigned)ldb);
        const __bf16* pan = &Bpanel[p & 1][0];
#pragma unroll
        for (int kk = 0; kk < KP; kk += 32) {
            v16bf a  = load_a_tile(Ab, lda, m0, p * KP + kk, lane);
            v16bf bt = load_b_panel_lds(pan, kk, lane);
            acc = __builtin_amdgcn_wmma_f32_16x16x32_bf16(false, a, false, bt,
                                                          (short)0, acc, false, false);
        }
        __builtin_amdgcn_s_wait_tensorcnt(0);   // next panel landed (no-op on waves 1..3)
        __syncthreads();
    }

    int col = n0 + (lane & 15);
    int hi8 = (lane >> 4) * 8;
#pragma unroll
    for (int i = 0; i < 8; ++i) {
        int row = m0 + i + hi8;
        float v = acc[i] + (bias ? bias[row] : 0.f);
        if (MODE == 0) {
            ((float*)D + (long)b * sD)[(long)row * ldd + col] = v;
        } else if (MODE == 1) {
            ((__bf16*)D + (long)b * sD)[(long)row * ldd + col] = f2bf(v);
        } else {
            ((__bf16*)D + (long)b * sD)[(long)col * ldd + row] = f2bf(v);
        }
    }
}

// ---------- channel softmax (over c, per (b,d) column of G[C,C]) ----------

__global__ __launch_bounds__(256)
void channel_softmax_kernel(const float* __restrict__ G, __bf16* __restrict__ Ach) {
    int bd = blockIdx.x;
    int b  = bd >> 8;
    int d  = bd & 255;
    int c  = threadIdx.x;
    const float* g = G + (long)b * Cc * Cc;
    float x = g[(long)c * Cc + d];

    __shared__ float red[256];
    red[c] = x; __syncthreads();
    for (int s = 128; s > 0; s >>= 1) {
        if (c < s) red[c] = fmaxf(red[c], red[c + s]);
        __syncthreads();
    }
    float mx = red[0]; __syncthreads();
    float ex = __expf(x - mx);
    red[c] = ex; __syncthreads();
    for (int s = 128; s > 0; s >>= 1) {
        if (c < s) red[c] += red[c + s];
        __syncthreads();
    }
    float sum = red[0];
    Ach[(long)b * Cc * Cc + (long)c * Cc + d] = f2bf(ex / sum);
}

// ---------- fused spatial attention (flash-style, softmax over batch) ----------
// 1 workgroup = one 16-column m-tile; wave w handles batch b=w.
// Loop n in steps of 32: S[b](32x16) via 4 WMMAs (K=64), LDS exchange,
// 4-way batch softmax, P repacked as WMMA-B, then O[c-tile] += V-tile * P.
__global__ __launch_bounds__(128)
void spatial_flash_kernel(const __bf16* __restrict__ qT,   // [B, N, QK]
                          const __bf16* __restrict__ kk,   // [B, QK, N]
                          const __bf16* __restrict__ vv,   // [B, C, N]
                          float* __restrict__ out) {       // [B, C, N] (+=)
    __shared__ float S[Bn][32][16];
    int lane = threadIdx.x & 31;
    int b    = threadIdx.x >> 5;
    int m0   = blockIdx.x * 16;
    int cl   = lane & 15;
    int hi   = lane >> 4;
    int hi8  = hi * 8;

    const __bf16* qTb = qT + (long)b * Nn * QKd;
    const __bf16* kb  = kk + (long)b * QKd * Nn;
    const __bf16* vb  = vv + (long)b * Cc * Nn;

    // k columns for this m-tile are loop-invariant: preload both K-step B operands
    v16bf bk0 = load_b_tile(kb, Nn, 0,  m0, lane);
    v16bf bk1 = load_b_tile(kb, Nn, 32, m0, lane);

    const v8f vzero = {0.f, 0.f, 0.f, 0.f, 0.f, 0.f, 0.f, 0.f};
    v8f accO[16];
#pragma unroll
    for (int t = 0; t < 16; ++t) accO[t] = vzero;

    for (int n0 = 0; n0 < Nn; n0 += 32) {
        // S tiles: rows n0..n0+31, cols m0..m0+15, K = QK = 64
        v16bf a00 = load_a_tile(qTb, QKd, n0,      0,  lane);
        v16bf a01 = load_a_tile(qTb, QKd, n0,      32, lane);
        v16bf a10 = load_a_tile(qTb, QKd, n0 + 16, 0,  lane);
        v16bf a11 = load_a_tile(qTb, QKd, n0 + 16, 32, lane);
        v8f s0 = vzero, s1 = vzero;
        s0 = __builtin_amdgcn_wmma_f32_16x16x32_bf16(false, a00, false, bk0, (short)0, s0, false, false);
        s0 = __builtin_amdgcn_wmma_f32_16x16x32_bf16(false, a01, false, bk1, (short)0, s0, false, false);
        s1 = __builtin_amdgcn_wmma_f32_16x16x32_bf16(false, a10, false, bk0, (short)0, s1, false, false);
        s1 = __builtin_amdgcn_wmma_f32_16x16x32_bf16(false, a11, false, bk1, (short)0, s1, false, false);

#pragma unroll
        for (int i = 0; i < 8; ++i) {
            S[b][i + hi8][cl]      = s0[i];
            S[b][16 + i + hi8][cl] = s1[i];
        }
        __syncthreads();

        // softmax across batch (4 values per (n,m)); gather P directly in B layout
        union { v16bf v; __bf16 e[16]; } pb;
        int kbase = hi * 16;
#pragma unroll
        for (int e = 0; e < 16; ++e) {
            int r = kbase + e;
            float x0 = S[0][r][cl], x1 = S[1][r][cl];
            float x2 = S[2][r][cl], x3 = S[3][r][cl];
            float mx = fmaxf(fmaxf(x0, x1), fmaxf(x2, x3));
            float e0 = __expf(x0 - mx), e1 = __expf(x1 - mx);
            float e2 = __expf(x2 - mx), e3 = __expf(x3 - mx);
            float sum = e0 + e1 + e2 + e3;
            float xb  = S[b][r][cl];
            pb.e[e] = f2bf(__expf(xb - mx) / sum);
        }
        __syncthreads();

        // O[c,m] += V[c, n0..n0+31] * P[n0..n0+31, m]  (all 256 channels)
#pragma unroll
        for (int t = 0; t < 16; ++t) {
            v16bf av = load_a_tile(vb, Nn, t * 16, n0, lane);
            accO[t] = __builtin_amdgcn_wmma_f32_16x16x32_bf16(false, av, false, pb.v,
                                                              (short)0, accO[t], false, false);
        }
    }

    float* ob = out + (long)b * Cc * Nn;
#pragma unroll
    for (int t = 0; t < 16; ++t) {
#pragma unroll
        for (int i = 0; i < 8; ++i) {
            int c = t * 16 + i + hi8;
            ob[(long)c * Nn + m0 + cl] += accO[t][i];
        }
    }
}

// ---------- host launcher ----------

extern "C" void kernel_launch(void* const* d_in, const int* in_sizes, int n_in,
                              void* d_out, int out_size, void* d_ws, size_t ws_size,
                              hipStream_t stream) {
    (void)in_sizes; (void)n_in; (void)out_size; (void)ws_size;

    const float* features = (const float*)d_in[0];
    const float* Wq = (const float*)d_in[1];
    const float* bq = (const float*)d_in[2];
    const float* Wk = (const float*)d_in[3];
    const float* bk = (const float*)d_in[4];
    const float* Wv = (const float*)d_in[5];
    const float* bv = (const float*)d_in[6];
    float* out = (float*)d_out;

    char* p = (char*)d_ws;
    auto alloc = [&](size_t bytes) {
        void* r = (void*)p;
        p += (bytes + 255) & ~(size_t)255;
        return r;
    };
    __bf16* x_bf  = (__bf16*)alloc((size_t)Bn * Cc * Nn * 2);
    __bf16* xT_bf = (__bf16*)alloc((size_t)Bn * Nn * Cc * 2);
    __bf16* Wq_bf = (__bf16*)alloc((size_t)QKd * Cc * 2);
    __bf16* Wk_bf = (__bf16*)alloc((size_t)QKd * Cc * 2);
    __bf16* Wv_bf = (__bf16*)alloc((size_t)Cc * Cc * 2);
    __bf16* qT_bf = (__bf16*)alloc((size_t)Bn * Nn * QKd * 2);
    __bf16* k_bf  = (__bf16*)alloc((size_t)Bn * QKd * Nn * 2);
    __bf16* v_bf  = (__bf16*)alloc((size_t)Bn * Cc * Nn * 2);
    float*  Gf    = (float*)alloc((size_t)Bn * Cc * Cc * 4);
    __bf16* Ach   = (__bf16*)alloc((size_t)Bn * Cc * Cc * 2);

    // 1. bf16 conversions
    {
        long total = (long)Bn * Cc * Nn;
        convert_features_kernel<<<dim3((unsigned)((total + 255) / 256)), 256, 0, stream>>>(
            features, x_bf, xT_bf);
        convert_f32_bf16_kernel<<<dim3((QKd * Cc + 255) / 256), 256, 0, stream>>>(Wq, Wq_bf, QKd * Cc);
        convert_f32_bf16_kernel<<<dim3((QKd * Cc + 255) / 256), 256, 0, stream>>>(Wk, Wk_bf, QKd * Cc);
        convert_f32_bf16_kernel<<<dim3((Cc * Cc + 255) / 256), 256, 0, stream>>>(Wv, Wv_bf, Cc * Cc);
    }

    // 2. projections
    // qT(b)[N,QK] = (Wq * x(b) + bq)^T
    gemm_bf16_wmma<2><<<dim3(Nn / 16, QKd / 64, Bn), 128, 0, stream>>>(
        Wq_bf, 0L, x_bf, (long)Cc * Nn, bq, qT_bf, (long)Nn * QKd,
        Cc, Cc, Nn, QKd);
    // k(b)[QK,N] = Wk * x(b) + bk
    gemm_bf16_wmma<1><<<dim3(Nn / 16, QKd / 64, Bn), 128, 0, stream>>>(
        Wk_bf, 0L, x_bf, (long)Cc * Nn, bk, k_bf, (long)QKd * Nn,
        Cc, Cc, Nn, Nn);
    // v(b)[C,N] = Wv * x(b) + bv
    gemm_bf16_wmma<1><<<dim3(Nn / 16, Cc / 64, Bn), 128, 0, stream>>>(
        Wv_bf, 0L, x_bf, (long)Cc * Nn, bv, v_bf, (long)Cc * Nn,
        Cc, Cc, Nn, Nn);

    // 3. channel attention
    // G(b)[C,C] = x(b) * x(b)^T
    gemm_bf16_wmma<0><<<dim3(Cc / 16, Cc / 64, Bn), 128, 0, stream>>>(
        x_bf, (long)Cc * Nn, xT_bf, (long)Nn * Cc, nullptr, Gf, (long)Cc * Cc,
        Nn, Nn, Cc, Cc);
    // softmax over c per column d
    channel_softmax_kernel<<<dim3(Bn * Cc), 256, 0, stream>>>(Gf, Ach);
    // out(b)[C,N] = Ach(b) * x(b)   (overwrites poisoned d_out)
    gemm_bf16_wmma<0><<<dim3(Nn / 16, Cc / 64, Bn), 128, 0, stream>>>(
        Ach, (long)Cc * Cc, x_bf, (long)Cc * Nn, nullptr, out, (long)Cc * Nn,
        Cc, Cc, Nn, Nn);

    // 4. fused spatial attention, accumulated on top of the channel term
    spatial_flash_kernel<<<dim3(Nn / 16), 128, 0, stream>>>(qT_bf, k_bf, v_bf, out);
}
